// DDAP_68058051772599
// MI455X (gfx1250) — matrix operand order repaired and checked by reading, not compiled, not run on hardware
//
#include <hip/hip_runtime.h>
#include <hip/hip_bf16.h>
#include <math.h>

#define NPTS 30000
#define KNB  16
#define DIN  64
#define DOUT 128
#define EPSV 1e-5f

typedef __attribute__((ext_vector_type(16))) _Float16 v16h;
typedef __attribute__((ext_vector_type(8)))  _Float16 v8h;
typedef __attribute__((ext_vector_type(8)))  float    v8f;

__device__ __forceinline__ float leaky_f(float x) { return x >= 0.f ? x : 0.2f * x; }

// ---------------------------------------------------------------------------
// Pre-swizzle a weight matrix [OUT x IN] into per-lane WMMA B-fragment order:
// dst[((tt*nChunks + c)*32 + lane)*16 + j] = W[out=tt*16+(lane&15)][in],
// in = c*32 + (lane<16 ? 0 : 8) + j + (j<8 ? 0 : 8), zero-padded past IN.
// ---------------------------------------------------------------------------
__global__ void swz_weights_kernel(const float* __restrict__ Wsrc,
                                   _Float16* __restrict__ dst,
                                   int IN, int nTiles, int nChunks) {
  int total = nTiles * nChunks * 32 * 16;
  for (int idx = blockIdx.x * blockDim.x + threadIdx.x; idx < total;
       idx += gridDim.x * blockDim.x) {
    int j    = idx & 15;
    int lane = (idx >> 4) & 31;
    int f    = idx >> 9;
    int c    = f % nChunks;
    int tt   = f / nChunks;
    int in   = c * 32 + ((lane < 16) ? 0 : 8) + j + ((j < 8) ? 0 : 8);
    int out  = tt * 16 + (lane & 15);
    dst[idx] = (_Float16)((in < IN) ? Wsrc[out * IN + in] : 0.f);
  }
}

// ---------------------------------------------------------------------------
// Batched GEMM via WMMA: Y[N,C] = A[N,128](f16) @ Bswz^T + bias.
// One wave per 16-row tile, 8 waves per block; B fragments staged in LDS.
// ---------------------------------------------------------------------------
__global__ void __launch_bounds__(256)
gemm_wmma_kernel(const _Float16* __restrict__ Arows,  // [N][128] f16 row-major
                 const _Float16* __restrict__ Bswz,   // nT*4*32 v16h fragments
                 const float* __restrict__ bias,
                 float* __restrict__ Y, int Nrows, int Ccols) {
  __shared__ v16h bs[8 * 4 * 32];                     // up to 32 KB
  const int t  = threadIdx.x;
  const int nT = Ccols >> 4;
  {
    const v16h* src = (const v16h*)Bswz;
    for (int i = t; i < nT * 4 * 32; i += 256) bs[i] = src[i];
  }
  __syncthreads();

  const int wave = t >> 5;
  const int ln   = t & 31;
  const int tile = blockIdx.x * 8 + wave;
  if (tile * 16 >= Nrows) return;                     // wave-uniform exit

  const int r    = ln & 15;
  const int off0 = (ln < 16) ? 0 : 8;
  const int mb   = (ln < 16) ? 0 : 8;
  const int row  = tile * 16 + r;
  const _Float16* ap = Arows + (size_t)row * 128;

  v16h afrag[4];
#pragma unroll
  for (int c = 0; c < 4; c++) {
    int kb = c * 32;
    v8h lo = *(const v8h*)(ap + kb + off0);
    v8h hi = *(const v8h*)(ap + kb + 16 + off0);
    v16h a;
#pragma unroll
    for (int j = 0; j < 8; j++) { a[j] = lo[j]; a[j + 8] = hi[j]; }
    afrag[c] = a;
  }

  for (int tt = 0; tt < nT; tt++) {
    v16h bfr[4];
#pragma unroll
    for (int c = 0; c < 4; c++) bfr[c] = bs[(tt * 4 + c) * 32 + ln];
    v8f acc = {};
#pragma unroll
    for (int c = 0; c < 4; c++)
      acc = __builtin_amdgcn_wmma_f32_16x16x32_f16(false, afrag[c], false,
                                                   bfr[c], (short)0, acc,
                                                   false, false);
    int col = tt * 16 + r;
    float bv = bias[col];
#pragma unroll
    for (int i = 0; i < 8; i++)
      Y[(size_t)(tile * 16 + mb + i) * Ccols + col] = acc[i] + bv;
  }
}

// ---------------------------------------------------------------------------
// gvol: per-segment max of ||xyz||
// ---------------------------------------------------------------------------
__global__ void gvol_part_kernel(const float* __restrict__ xyz,
                                 const int* __restrict__ lengths,
                                 float* __restrict__ part, int n) {
  __shared__ float sm0[256], sm1[256];
  const int L0 = lengths[0];
  float m0 = 0.f, m1 = 0.f;
  for (int i = blockIdx.x * blockDim.x + threadIdx.x; i < n;
       i += gridDim.x * blockDim.x) {
    float x = xyz[3 * i], y = xyz[3 * i + 1], z = xyz[3 * i + 2];
    float gd = sqrtf(x * x + y * y + z * z);
    if (i < L0) m0 = fmaxf(m0, gd); else m1 = fmaxf(m1, gd);
  }
  sm0[threadIdx.x] = m0; sm1[threadIdx.x] = m1;
  __syncthreads();
  for (int s = 128; s > 0; s >>= 1) {
    if (threadIdx.x < s) {
      sm0[threadIdx.x] = fmaxf(sm0[threadIdx.x], sm0[threadIdx.x + s]);
      sm1[threadIdx.x] = fmaxf(sm1[threadIdx.x], sm1[threadIdx.x + s]);
    }
    __syncthreads();
  }
  if (threadIdx.x == 0) {
    part[2 * blockIdx.x]     = sm0[0];
    part[2 * blockIdx.x + 1] = sm1[0];
  }
}

__global__ void gvol_fin_kernel(const float* __restrict__ part,
                                float* __restrict__ gvol, int nb) {
  if (threadIdx.x == 0 && blockIdx.x == 0) {
    float m0 = 0.f, m1 = 0.f;
    for (int i = 0; i < nb; i++) {
      m0 = fmaxf(m0, part[2 * i]);
      m1 = fmaxf(m1, part[2 * i + 1]);
    }
    gvol[0] = m0; gvol[1] = m1;
  }
}

// ---------------------------------------------------------------------------
// Channel-wise mean/var over N (InstanceNorm axis=0)
// ---------------------------------------------------------------------------
__global__ void chan_reduce_kernel(const float* __restrict__ y,
                                   float* __restrict__ part, int n, int C) {
  __shared__ float ss[256], qq[256];
  const int t = threadIdx.x;
  const int c = t % C;
  const int g = t / C;
  const int G = blockDim.x / C;
  float s = 0.f, q = 0.f;
  for (int i = blockIdx.x * G + g; i < n; i += gridDim.x * G) {
    float v = y[(size_t)i * C + c];
    s += v; q += v * v;
  }
  ss[t] = s; qq[t] = q;
  __syncthreads();
  if (g == 0) {
    for (int k = 1; k < G; k++) { s += ss[k * C + c]; q += qq[k * C + c]; }
    part[(size_t)blockIdx.x * 2 * C + c]     = s;
    part[(size_t)blockIdx.x * 2 * C + C + c] = q;
  }
}

__global__ void chan_fin_kernel(const float* __restrict__ part,
                                float* __restrict__ mean,
                                float* __restrict__ rstd,
                                int nb, int C, int n) {
  int c = threadIdx.x;
  if (c >= C) return;
  float s = 0.f, q = 0.f;
  for (int b = 0; b < nb; b++) {
    s += part[(size_t)b * 2 * C + c];
    q += part[(size_t)b * 2 * C + C + c];
  }
  float m = s / (float)n;
  float v = q / (float)n - m * m;
  mean[c] = m;
  rstd[c] = rsqrtf(v + EPSV);
}

__global__ void norm_leaky_kernel(const float* __restrict__ y,
                                  const float* __restrict__ mean,
                                  const float* __restrict__ rstd,
                                  float* __restrict__ out, int total, int C) {
  int i = blockIdx.x * blockDim.x + threadIdx.x;
  if (i >= total) return;
  int c = i % C;
  float v = (y[i] - mean[c]) * rstd[c];
  out[i] = leaky_f(v);
}

// ---------------------------------------------------------------------------
// Stage 1: one wave (32 lanes) per point; emits pooled s1 (f16 [N][128]).
// ---------------------------------------------------------------------------
__global__ void __launch_bounds__(32)
stage1_kernel(const float* __restrict__ xyz, const float* __restrict__ feat,
              const int* __restrict__ nidx, const int* __restrict__ lengths,
              const float* __restrict__ W1, const float* __restrict__ b1,
              const _Float16* __restrict__ wfc1_swz,
              const float* __restrict__ gvol,
              float* __restrict__ gdis_ws, _Float16* __restrict__ lr1_ws,
              _Float16* __restrict__ s1_ws, float* __restrict__ lgr_out) {
  __shared__ v16h  wfcs[8 * 5 * 32];   // pre-swizzled Wfc1 B fragments (40 KB)
  __shared__ float cc[16][164];
  __shared__ float rep[16][12];
  __shared__ float ssum[128];

  const int n = blockIdx.x;
  const int t = threadIdx.x;

  {
    const v16h* src = (const v16h*)wfc1_swz;
    for (int i = t; i < 8 * 5 * 32; i += 32) wfcs[i] = src[i];
  }
  for (int idx = t; idx < 16 * 164; idx += 32) ((float*)cc)[idx] = 0.f;
  __syncthreads();

  const float cx = xyz[3 * n], cy = xyz[3 * n + 1], cz = xyz[3 * n + 2];

  int nbi = 0;
  float nx = 0.f, ny = 0.f, nz = 0.f;
  if (t < KNB) {
    nbi = nidx[n * KNB + t];
    nx = xyz[3 * nbi]; ny = xyz[3 * nbi + 1]; nz = xyz[3 * nbi + 2];
  }
  float sx = nx, sy = ny, sz = nz;
  for (int w = 1; w < 16; w <<= 1) {
    sx += __shfl_xor(sx, w, 32);
    sy += __shfl_xor(sy, w, 32);
    sz += __shfl_xor(sz, w, 32);
  }
  sx = __shfl(sx, 0, 32); sy = __shfl(sy, 0, 32); sz = __shfl(sz, 0, 32);
  const float dirx = cx - sx / 16.f, diry = cy - sy / 16.f, dirz = cz - sz / 16.f;
  const float dir_alpha = atan2f(diry, dirx);
  const float dir_beta  = atan2f(dirz, sqrtf(dirx * dirx + diry * diry));

  float rdis = 0.f;
  if (t < KNB) {
    float rx = cx - nx, ry = cy - ny, rz = cz - nz;
    float ra = atan2f(ry, rx);
    float rb = atan2f(rz, sqrtf(rx * rx + ry * ry));
    rdis = sqrtf(rx * rx + ry * ry + rz * rz);
    rep[t][0] = ra - dir_alpha;
    rep[t][1] = rb - dir_beta;
    rep[t][2] = rdis;
    rep[t][3] = cx; rep[t][4] = cy; rep[t][5] = cz;
    rep[t][6] = nx; rep[t][7] = ny; rep[t][8] = nz;
    float gd = expf(-rdis);
    cc[t][0] = gd;
    gdis_ws[n * KNB + t] = gd;
  }
  float md = rdis;
  for (int w = 1; w < 16; w <<= 1) md = fmaxf(md, __shfl_xor(md, w, 32));
  if (t == 0) {
    int L0 = lengths[0];
    float gv = gvol[(n < L0) ? 0 : 1];
    lgr_out[n] = (md * md * md) / gv;
  }

  if (t < KNB) {
    float acc = 0.f;
    for (int d = 0; d < DIN; d++) {
      float fv = feat[(size_t)nbi * DIN + d];
      cc[t][2 + d] = fv;
      acc += fabsf(feat[(size_t)n * DIN + d] - fv);
    }
    cc[t][1] = 0.1f * expf(-acc / 64.f);
  }
  __syncthreads();

  // conv1 (9-dot, VALU)
  for (int idx = t; idx < 16 * 64; idx += 32) {
    int k = idx >> 6, d = idx & 63;
    float s = b1[d];
#pragma unroll
    for (int j = 0; j < 9; j++) s += rep[k][j] * W1[d * 9 + j];
    cc[k][66 + d] = s;
  }
  __syncthreads();

  // InstanceNorm over K + leaky; persist lr1 (f16)
  for (int d = t; d < 64; d += 32) {
    float s = 0.f, q = 0.f;
    for (int k = 0; k < 16; k++) { float v = cc[k][66 + d]; s += v; q += v * v; }
    float m  = s / 16.f;
    float rv = rsqrtf(q / 16.f - m * m + EPSV);
    for (int k = 0; k < 16; k++) {
      float v = leaky_f((cc[k][66 + d] - m) * rv);
      cc[k][66 + d] = v;
      lr1_ws[(size_t)n * 16 * 64 + k * 64 + d] = (_Float16)v;
    }
  }
  __syncthreads();

  const int r    = t & 15;
  const int off0 = (t < 16) ? 0 : 8;
  const int mb   = (t < 16) ? 0 : 8;
  v16h afrag[5];
#pragma unroll
  for (int c = 0; c < 5; c++) {
    int kb = c * 32;
    v16h a;
#pragma unroll
    for (int j = 0; j < 8; j++) {
      a[j]     = (_Float16)cc[r][kb + off0 + j];
      a[j + 8] = (_Float16)cc[r][kb + 16 + off0 + j];
    }
    afrag[c] = a;
  }

  for (int tt = 0; tt < 8; tt++) {
    v16h bfr[5];
#pragma unroll
    for (int c = 0; c < 5; c++) bfr[c] = wfcs[(tt * 5 + c) * 32 + t];
    v8f acc = {};
#pragma unroll
    for (int c = 0; c < 5; c++)
      acc = __builtin_amdgcn_wmma_f32_16x16x32_f16(false, afrag[c], false,
                                                   bfr[c], (short)0, acc,
                                                   false, false);
    int col = tt * 16 + r;
    float mx = acc[0];
#pragma unroll
    for (int i = 1; i < 8; i++) mx = fmaxf(mx, acc[i]);
    mx = fmaxf(mx, __shfl_xor(mx, 16, 32));
    float e[8], sum = 0.f;
#pragma unroll
    for (int i = 0; i < 8; i++) { e[i] = expf(acc[i] - mx); sum += e[i]; }
    sum += __shfl_xor(sum, 16, 32);
    float s = 0.f;
#pragma unroll
    for (int i = 0; i < 8; i++) s += (e[i] / sum) * cc[mb + i][2 + col];
    s += __shfl_xor(s, 16, 32);
    if (t < 16) ssum[col] = s;
  }
  __syncthreads();

  // pooled vector out (f16) -> GEMM kernel does s1 @ Wm1^T + bm1
  for (int d = t; d < 128; d += 32)
    s1_ws[(size_t)n * 128 + d] = (_Float16)ssum[d];
}

// ---------------------------------------------------------------------------
// Stage 2: one wave per point; emits pooled s2 (f16 [N][128]).
// ---------------------------------------------------------------------------
__global__ void __launch_bounds__(32)
stage2_kernel(const int* __restrict__ nidx,
              const _Float16* __restrict__ w2_swz, const float* __restrict__ b2,
              const _Float16* __restrict__ wfc2_swz,
              const float* __restrict__ flc_ws, const float* __restrict__ gdis_ws,
              const _Float16* __restrict__ lr1_ws, _Float16* __restrict__ s2_ws) {
  __shared__ v16h     wfcs[8 * 5 * 32];  // Wfc2 fragments (40 KB)
  __shared__ v16h     w2s[4 * 2 * 32];   // W2 fragments (8 KB)
  __shared__ _Float16 lrb[16][64];
  __shared__ float    cc[16][164];
  __shared__ float    ssum[128];

  const int n = blockIdx.x;
  const int t = threadIdx.x;

  {
    const v16h* src = (const v16h*)wfc2_swz;
    for (int i = t; i < 8 * 5 * 32; i += 32) wfcs[i] = src[i];
    const v16h* src2 = (const v16h*)w2_swz;
    for (int i = t; i < 4 * 2 * 32; i += 32) w2s[i] = src2[i];
    const v16h* src3 = (const v16h*)(lr1_ws + (size_t)n * 1024);
    v16h* dst3 = (v16h*)lrb;
    for (int i = t; i < 64; i += 32) dst3[i] = src3[i];
  }
  for (int idx = t; idx < 16 * 164; idx += 32) ((float*)cc)[idx] = 0.f;
  __syncthreads();

  const int r    = t & 15;
  const int off0 = (t < 16) ? 0 : 8;
  const int mb   = (t < 16) ? 0 : 8;

  // conv2 via WMMA: [16,64] @ [64,64]
  v16h a2[2];
#pragma unroll
  for (int c = 0; c < 2; c++) {
    int kb = c * 32;
    v16h a;
#pragma unroll
    for (int j = 0; j < 8; j++) {
      a[j]     = lrb[r][kb + off0 + j];
      a[j + 8] = lrb[r][kb + 16 + off0 + j];
    }
    a2[c] = a;
  }
  for (int tt = 0; tt < 4; tt++) {
    v16h bfr[2];
#pragma unroll
    for (int c = 0; c < 2; c++) bfr[c] = w2s[(tt * 2 + c) * 32 + t];
    v8f acc = {};
#pragma unroll
    for (int c = 0; c < 2; c++)
      acc = __builtin_amdgcn_wmma_f32_16x16x32_f16(false, a2[c], false, bfr[c],
                                                   (short)0, acc, false, false);
    int col = tt * 16 + r;
#pragma unroll
    for (int i = 0; i < 8; i++) cc[mb + i][66 + col] = acc[i] + b2[col];
  }
  __syncthreads();

  for (int d = t; d < 64; d += 32) {
    float s = 0.f, q = 0.f;
    for (int k = 0; k < 16; k++) { float v = cc[k][66 + d]; s += v; q += v * v; }
    float m  = s / 16.f;
    float rv = rsqrtf(q / 16.f - m * m + EPSV);
    for (int k = 0; k < 16; k++) cc[k][66 + d] = leaky_f((cc[k][66 + d] - m) * rv);
  }

  if (t < KNB) {
    int nbi = nidx[n * KNB + t];
    float acc = 0.f;
    for (int d = 0; d < 64; d++) {
      float fv = flc_ws[(size_t)nbi * 64 + d];
      cc[t][2 + d] = fv;
      acc += fabsf(flc_ws[(size_t)n * 64 + d] - fv);
    }
    cc[t][1] = 0.1f * expf(-acc / 64.f);
    cc[t][0] = gdis_ws[n * KNB + t];
  }
  __syncthreads();

  v16h afrag[5];
#pragma unroll
  for (int c = 0; c < 5; c++) {
    int kb = c * 32;
    v16h a;
#pragma unroll
    for (int j = 0; j < 8; j++) {
      a[j]     = (_Float16)cc[r][kb + off0 + j];
      a[j + 8] = (_Float16)cc[r][kb + 16 + off0 + j];
    }
    afrag[c] = a;
  }
  for (int tt = 0; tt < 8; tt++) {
    v16h bfr[5];
#pragma unroll
    for (int c = 0; c < 5; c++) bfr[c] = wfcs[(tt * 5 + c) * 32 + t];
    v8f acc = {};
#pragma unroll
    for (int c = 0; c < 5; c++)
      acc = __builtin_amdgcn_wmma_f32_16x16x32_f16(false, afrag[c], false,
                                                   bfr[c], (short)0, acc,
                                                   false, false);
    int col = tt * 16 + r;
    float mx = acc[0];
#pragma unroll
    for (int i = 1; i < 8; i++) mx = fmaxf(mx, acc[i]);
    mx = fmaxf(mx, __shfl_xor(mx, 16, 32));
    float e[8], sum = 0.f;
#pragma unroll
    for (int i = 0; i < 8; i++) { e[i] = expf(acc[i] - mx); sum += e[i]; }
    sum += __shfl_xor(sum, 16, 32);
    float s = 0.f;
#pragma unroll
    for (int i = 0; i < 8; i++) s += (e[i] / sum) * cc[mb + i][2 + col];
    s += __shfl_xor(s, 16, 32);
    if (t < 16) ssum[col] = s;
  }
  __syncthreads();

  for (int d = t; d < 128; d += 32)
    s2_ws[(size_t)n * 128 + d] = (_Float16)ssum[d];
}

// ---------------------------------------------------------------------------
// Launch
// ---------------------------------------------------------------------------
extern "C" void kernel_launch(void* const* d_in, const int* in_sizes, int n_in,
                              void* d_out, int out_size, void* d_ws, size_t ws_size,
                              hipStream_t stream) {
  const int N = NPTS;
  const float* xyz   = (const float*)d_in[0];
  const float* feat  = (const float*)d_in[1];
  const int*   nidx  = (const int*)d_in[2];
  const int*   lens  = (const int*)d_in[3];
  const float* W1    = (const float*)d_in[4];
  const float* b1    = (const float*)d_in[5];
  const float* W2    = (const float*)d_in[6];
  const float* b2    = (const float*)d_in[7];
  const float* Wfc1  = (const float*)d_in[8];
  const float* Wfc2  = (const float*)d_in[9];
  const float* Wm1   = (const float*)d_in[10];
  const float* bm1   = (const float*)d_in[11];
  const float* Wm2   = (const float*)d_in[12];
  const float* bm2   = (const float*)d_in[13];
  float* out = (float*)d_out;

  float* W = (float*)d_ws;
  const size_t off_gpart = 0;                           // 128
  const size_t off_gvol  = 128;                         // 2
  const size_t off_mean1 = 130;                         // 64
  const size_t off_rstd1 = 194;                         // 64
  const size_t off_mean2 = 258;                         // 128
  const size_t off_rstd2 = 386;                         // 128
  const size_t off_red   = 512;                         // 30720
  const size_t off_wfc1s = 31232;                       // 10240 f
  const size_t off_wfc2s = off_wfc1s + 10240;           // 10240 f
  const size_t off_w2s   = off_wfc2s + 10240;           // 2048 f
  const size_t off_wm1s  = off_w2s + 2048;              // 2048 f (4096 halfs)
  const size_t off_wm2s  = off_wm1s + 2048;             // 8192 f (16384 halfs)
  const size_t off_gdis  = off_wm2s + 8192;             // N*16
  const size_t off_y1    = off_gdis + (size_t)N * 16;   // N*64
  const size_t off_flc   = off_y1 + (size_t)N * 64;     // N*64
  const size_t off_y2    = off_flc + (size_t)N * 64;    // N*128
  const size_t off_s1h   = off_y2 + (size_t)N * 128;    // N*128 halfs = N*64 f
  const size_t off_s2h   = off_s1h + (size_t)N * 64;    // N*64 f
  const size_t off_lr1h  = off_s2h + (size_t)N * 64;    // N*1024 halfs
  _Float16* wfc1s = (_Float16*)(W + off_wfc1s);
  _Float16* wfc2s = (_Float16*)(W + off_wfc2s);
  _Float16* w2s   = (_Float16*)(W + off_w2s);
  _Float16* wm1s  = (_Float16*)(W + off_wm1s);
  _Float16* wm2s  = (_Float16*)(W + off_wm2s);
  _Float16* s1h   = (_Float16*)(W + off_s1h);
  _Float16* s2h   = (_Float16*)(W + off_s2h);
  _Float16* lr1h  = (_Float16*)(W + off_lr1h);

  // weight pre-swizzle into WMMA B-fragment order
  swz_weights_kernel<<<40, 256, 0, stream>>>(Wfc1, wfc1s, 130, 8, 5);
  swz_weights_kernel<<<40, 256, 0, stream>>>(Wfc2, wfc2s, 130, 8, 5);
  swz_weights_kernel<<<16, 256, 0, stream>>>(W2, w2s, 64, 4, 2);
  swz_weights_kernel<<<16, 256, 0, stream>>>(Wm1, wm1s, 128, 4, 4);
  swz_weights_kernel<<<32, 256, 0, stream>>>(Wm2, wm2s, 128, 8, 4);

  // gvol (needed by stage1 for lg_ratio)
  gvol_part_kernel<<<64, 256, 0, stream>>>(xyz, lens, W + off_gpart, N);
  gvol_fin_kernel<<<1, 32, 0, stream>>>(W + off_gpart, W + off_gvol, 64);

  // stage 1 -> s1 (f16)
  stage1_kernel<<<N, 32, 0, stream>>>(xyz, feat, nidx, lens, W1, b1, wfc1s,
                                      W + off_gvol, W + off_gdis, lr1h, s1h,
                                      out + (size_t)N * 128);

  // y1 = s1 @ Wm1^T + bm1 (batched WMMA GEMM), then InstanceNorm over N
  const int nTiles = N / 16;                    // 1875
  gemm_wmma_kernel<<<(nTiles + 7) / 8, 256, 0, stream>>>(s1h, wm1s, bm1,
                                                         W + off_y1, N, 64);
  chan_reduce_kernel<<<120, 256, 0, stream>>>(W + off_y1, W + off_red, N, 64);
  chan_fin_kernel<<<1, 64, 0, stream>>>(W + off_red, W + off_mean1,
                                        W + off_rstd1, 120, 64, N);
  norm_leaky_kernel<<<(N * 64 + 255) / 256, 256, 0, stream>>>(
      W + off_y1, W + off_mean1, W + off_rstd1, W + off_flc, N * 64, 64);

  // stage 2 -> s2 (f16)
  stage2_kernel<<<N, 32, 0, stream>>>(nidx, w2s, b2, wfc2s, W + off_flc,
                                      W + off_gdis, lr1h, s2h);

  // y2 = s2 @ Wm2^T + bm2, then InstanceNorm over N -> final output
  gemm_wmma_kernel<<<(nTiles + 7) / 8, 256, 0, stream>>>(s2h, wm2s, bm2,
                                                         W + off_y2, N, 128);
  chan_reduce_kernel<<<120, 256, 0, stream>>>(W + off_y2, W + off_red, N, 128);
  chan_fin_kernel<<<1, 128, 0, stream>>>(W + off_red, W + off_mean2,
                                         W + off_rstd2, 120, 128, N);
  norm_leaky_kernel<<<(N * 128 + 255) / 256, 256, 0, stream>>>(
      W + off_y2, W + off_mean2, W + off_rstd2, out, N * 128, 128);
}
